// UniqueEq2Net_40355512713224
// MI455X (gfx1250) — compile-verified, matrix operand order checked by reading, not run on hardware
//
#include <hip/hip_runtime.h>
#include <hip/hip_bf16.h>

// ---------------------------------------------------------------------------
// CDNA5 (gfx1250, wave32) implementation of UniqueEq2Net.
// Key algebraic simplification: P[b,d] = v v^T is rank-1 symmetric, so all 15
// Maron 2->2 basis ops reduce to functions of v, S=sum(v), t=sum(v^2).
// Heavy lifting uses v_wmma_f32_16x16x32_bf16 with f32 accumulation.
// LDS staging in eq_kernel uses the Tensor Data Mover (6-arg builtin).
// ---------------------------------------------------------------------------

typedef __attribute__((ext_vector_type(16))) __bf16 bf16x16;
typedef __attribute__((ext_vector_type(8)))  __bf16 bf16x8;
typedef __attribute__((ext_vector_type(8)))  float  f32x8;
typedef __attribute__((ext_vector_type(4)))  unsigned int u32x4;
typedef __attribute__((ext_vector_type(8)))  int i32x8;
typedef __attribute__((ext_vector_type(4)))  int i32x4;

#define DEV __device__ __forceinline__

// Build a 16x32 bf16 WMMA fragment from a row pointer (contiguous in K).
// CDNA5 wave32 layout: lane l, g=l>>4; halves 0..7 -> K = off+g*8+e,
// halves 8..15 -> K = off+16+g*8+e.
DEV bf16x16 mk_frag(const __bf16* rowp, int off) {
  bf16x8 lo = *(const bf16x8*)(rowp + off);
  bf16x8 hi = *(const bf16x8*)(rowp + off + 16);
  bf16x16 r;
#pragma unroll
  for (int e = 0; e < 8; ++e) { r[e] = lo[e]; r[e + 8] = hi[e]; }
  return r;
}

// ---------------------------------------------------------------------------
// conv1: (1024,1,28,28) -> relu -> (1024,32,12,12), 5x5 stride 2 VALID
// ---------------------------------------------------------------------------
__global__ void conv1_kernel(const float* __restrict__ x, const float* __restrict__ w,
                             const float* __restrict__ bias, float* __restrict__ y1) {
  int idx = blockIdx.x * 256 + threadIdx.x;
  if (idx >= 1024 * 32 * 144) return;
  int ox = idx % 12, oy = (idx / 12) % 12, c = (idx / 144) % 32, n = idx / 4608;
  const float* xp = x + n * 784 + (oy * 2) * 28 + (ox * 2);
  const float* wp = w + c * 25;
  float acc = bias[c];
#pragma unroll
  for (int ky = 0; ky < 5; ++ky)
#pragma unroll
    for (int kx = 0; kx < 5; ++kx)
      acc += xp[ky * 28 + kx] * wp[ky * 5 + kx];
  y1[idx] = fmaxf(acc, 0.0f);
}

// ---------------------------------------------------------------------------
// conv2: (1024,32,12,12) -> relu -> flattened (1024,1024) stored as bf16
// ---------------------------------------------------------------------------
__global__ void conv2_kernel(const float* __restrict__ y1, const float* __restrict__ w,
                             const float* __restrict__ bias, __bf16* __restrict__ act2) {
  int idx = blockIdx.x * 256 + threadIdx.x;
  if (idx >= 1024 * 64 * 16) return;
  int ox = idx % 4, oy = (idx / 4) % 4, c2 = (idx / 16) % 64, n = idx / 1024;
  const float* yp = y1 + n * 4608;
  const float* wp = w + c2 * 800;
  float acc = bias[c2];
  for (int ci = 0; ci < 32; ++ci) {
    const float* ypp = yp + ci * 144 + (oy * 2) * 12 + (ox * 2);
    const float* wpp = wp + ci * 25;
#pragma unroll
    for (int ky = 0; ky < 5; ++ky)
#pragma unroll
      for (int kx = 0; kx < 5; ++kx)
        acc += ypp[ky * 12 + kx] * wpp[ky * 5 + kx];
  }
  act2[n * 1024 + c2 * 16 + oy * 4 + ox] = (__bf16)fmaxf(acc, 0.0f);
}

// ---------------------------------------------------------------------------
// f32 -> bf16 conversion
// ---------------------------------------------------------------------------
__global__ void cvt_bf16_kernel(const float* __restrict__ in, __bf16* __restrict__ out, int n) {
  int t = blockIdx.x * 256 + threadIdx.x;
  if (t < n) out[t] = (__bf16)in[t];
}

// ---------------------------------------------------------------------------
// Coefficient preprocessing: fold the 15 basis ops into per-term weights.
// coeffs: (128 d, 128 s, 15 p). m = 64.
// ---------------------------------------------------------------------------
__global__ void prep_coeff_kernel(const float* __restrict__ coeffs,
                                  __bf16* __restrict__ wmT,   // (128 s,128 d): c10+c11
                                  __bf16* __restrict__ WcT,   // (384,256) transposed weights
                                  float* __restrict__ wC1, float* __restrict__ wC2,
                                  float* __restrict__ wE1, float* __restrict__ wE2) {
  int t = blockIdx.x * 256 + threadIdx.x;   // 16384 = 128*128
  if (t >= 16384) return;
  int d = t >> 7, s = t & 127;
  const float* c = coeffs + (size_t)(d * 128 + s) * 15;
  const float inv_m = 1.0f / 64.0f, inv_m2 = inv_m * inv_m;
  wmT[s * 128 + d] = (__bf16)(c[9] + c[10]);                 // ops 10,11: v_i v_j
  // X = [S*v | v^2] (K=256);   columns: [A | B | D] (N=384)
  WcT[(s)       * 256 + d]       = (__bf16)((c[5] + c[6]) * inv_m); // A: S*v part
  WcT[(s)       * 256 + 128 + d] = (__bf16)(c[11]);                 // A: v^2 part
  WcT[(128 + s) * 256 + d]       = (__bf16)((c[7] + c[8]) * inv_m); // B: S*v part
  WcT[(128 + s) * 256 + 128 + d] = (__bf16)(c[12]);                 // B: v^2 part
  WcT[(256 + s) * 256 + d]       = (__bf16)((c[2] + c[3]) * inv_m); // D: S*v part
  WcT[(256 + s) * 256 + 128 + d] = (__bf16)(c[0]);                  // D: v^2 part
  wC1[d * 128 + s] = c[13] * inv_m;   // op14: t/m (everywhere)
  wC2[d * 128 + s] = c[14] * inv_m2;  // op15: S^2/m^2 (everywhere)
  wE1[d * 128 + s] = c[1]  * inv_m;   // op2:  t/m (diag)
  wE2[d * 128 + s] = c[4]  * inv_m2;  // op5:  S^2/m^2 (diag)
}

// ---------------------------------------------------------------------------
// Generic bf16 WMMA GEMM:  D[MxN] = A[MxK] * Bt[NxK]^T (+ bias per column)
// One wave per 16x16 output tile.  A and Bt are row-major, contiguous in K.
// ---------------------------------------------------------------------------
__global__ void gemm_bf16_wmma(const __bf16* __restrict__ A, int lda,
                               const __bf16* __restrict__ Bt, int ldb,
                               const float* __restrict__ bias,
                               float* __restrict__ D, int ldd,
                               __bf16* __restrict__ Dbf, int K) {
  int l = threadIdx.x & 31;
  int g = l >> 4, mr = l & 15;
  const __bf16* arow = A  + (size_t)(blockIdx.x * 16 + mr) * lda;
  const __bf16* brow = Bt + (size_t)(blockIdx.y * 16 + mr) * ldb;
  f32x8 acc = {};
  for (int kc = 0; kc < K; kc += 32) {
    bf16x16 a = mk_frag(arow, kc + g * 8);
    bf16x16 b = mk_frag(brow, kc + g * 8);
    acc = __builtin_amdgcn_wmma_f32_16x16x32_bf16(false, a, false, b, (short)0, acc,
                                                  false, false);
  }
#pragma unroll
  for (int v = 0; v < 8; ++v) {
    int row = blockIdx.x * 16 + v + 8 * g;   // C/D layout: M = v + 8*(lane>=16)
    int col = blockIdx.y * 16 + mr;          // N = lane & 15
    float val = acc[v] + (bias ? bias[col] : 0.0f);
    D[(size_t)row * ldd + col] = val;
    if (Dbf) Dbf[(size_t)row * ldd + col] = (__bf16)val;
  }
}

// ---------------------------------------------------------------------------
// Per-(b,d) stats: S = sum_i feat, t = sum_i feat^2
// ---------------------------------------------------------------------------
__global__ void stats_kernel(const float* __restrict__ featf,
                             float* __restrict__ Sbuf, float* __restrict__ Tbuf) {
  int t = blockIdx.x * 256 + threadIdx.x;   // 2048 = 16*128
  if (t >= 2048) return;
  int b = t >> 7, d = t & 127;
  float s = 0.0f, q = 0.0f;
  for (int i = 0; i < 64; ++i) {
    float v = featf[(size_t)((b << 6) + i) * 128 + d];
    s += v; q += v * v;
  }
  Sbuf[t] = s; Tbuf[t] = q;
}

// X[(b,i), 0:128] = S*v,  X[(b,i), 128:256] = v^2   (bf16, K for A/B/D GEMM)
__global__ void buildX_kernel(const float* __restrict__ featf, const float* __restrict__ Sbuf,
                              __bf16* __restrict__ X) {
  int t = blockIdx.x * 256 + threadIdx.x;   // 131072 = 1024*128
  if (t >= 131072) return;
  int r = t >> 7, d = t & 127;
  int b = r >> 6;
  float v = featf[(size_t)r * 128 + d];
  X[(size_t)r * 256 + d]       = (__bf16)(Sbuf[b * 128 + d] * v);
  X[(size_t)r * 256 + 128 + d] = (__bf16)(v * v);
}

// C[b,s], E[b,s]: small K=128 dots plus biases
__global__ void ce_kernel(const float* __restrict__ Sbuf, const float* __restrict__ Tbuf,
                          const float* __restrict__ wC1, const float* __restrict__ wC2,
                          const float* __restrict__ wE1, const float* __restrict__ wE2,
                          const float* __restrict__ eqb, const float* __restrict__ dgb,
                          float* __restrict__ Cb, float* __restrict__ Eb) {
  int t = blockIdx.x * 256 + threadIdx.x;   // 2048 = 16*128
  if (t >= 2048) return;
  int b = t >> 7, s = t & 127;
  float ac = 0.0f, ae = 0.0f;
  for (int d = 0; d < 128; ++d) {
    float Sv = Sbuf[b * 128 + d], Tv = Tbuf[b * 128 + d];
    float S2 = Sv * Sv;
    ac += wC1[d * 128 + s] * Tv + wC2[d * 128 + s] * S2;
    ae += wE1[d * 128 + s] * Tv + wE2[d * 128 + s] * S2;
  }
  Cb[t] = ac + eqb[s];
  Eb[t] = ae + dgb[s];
}

// ---------------------------------------------------------------------------
// Main equivariant kernel: per (b,s) compute M = (V .* w_s) V^T via WMMA,
// add A_i + B_j + C + diag(D_i + E), relu, and mean over the 64x64 tile.
// One 256-thread block per (b, group of 8 s); feat tile staged in LDS via the
// Tensor Data Mover (falls back to vector copy if the builtin is absent).
// ---------------------------------------------------------------------------
__global__ void eq_kernel(const __bf16* __restrict__ featb, const __bf16* __restrict__ wmT,
                          const float* __restrict__ ABD, const float* __restrict__ Cb,
                          const float* __restrict__ Eb, float* __restrict__ meanb) {
  __shared__ __align__(16) __bf16 sV[64 * 128];
  int b = blockIdx.x;
  int lane = threadIdx.x & 31;
  int wave = threadIdx.x >> 5;
  int s = blockIdx.y * 8 + wave;
  int g = lane >> 4, mr = lane & 15;

#if defined(__HIP_DEVICE_COMPILE__) && __has_builtin(__builtin_amdgcn_tensor_load_to_lds)
  // TDM: DMA the 64x128 bf16 feat tile (16KB, contiguous) into LDS as a 1-D
  // tensor of 8192 2-byte elements.  One descriptor, issued by wave 0 only.
  if (threadIdx.x < 32) {
    unsigned long long ga = (unsigned long long)(featb + (size_t)b * 8192);
    u32x4 g0;
    g0[0] = 1u;                                   // count=1, user mode
    g0[1] = (unsigned)(size_t)sV;                 // lds_addr (bytes)
    g0[2] = (unsigned)(ga & 0xFFFFFFFFull);       // global_addr[31:0]
    g0[3] = (unsigned)((ga >> 32) & 0x1FFFFFFull) // global_addr[56:32]
            | (2u << 30);                         // type=2 ("image")
    i32x8 g1;
    g1[0] = (int)(1u << 16);            // data_size=1 (2B); no multicast/pad/iter
    g1[1] = (int)(8192u << 16);         // tensor_dim0[15:0]=8192 at bits 63:48
    g1[2] = (int)(1u << 16);            // tensor_dim0[31:16]=0; tensor_dim1=1
    g1[3] = (int)(8192u << 16);         // tensor_dim1[31:16]=0; tile_dim0=8192
    g1[4] = 0;                          // tile_dim1=0 (unused), tile_dim2=0
    g1[5] = 8192;                       // tensor_dim0_stride[31:0]
    g1[6] = 0;                          // stride hi / tensor_dim1_stride lo
    g1[7] = 0;
    i32x4 z4 = {0, 0, 0, 0};
    i32x8 z8 = {0, 0, 0, 0, 0, 0, 0, 0};
    __builtin_amdgcn_tensor_load_to_lds(g0, g1, z4, z4, z8, 0);
#if __has_builtin(__builtin_amdgcn_s_wait_tensorcnt)
    __builtin_amdgcn_s_wait_tensorcnt(0);
#endif
  }
#else
  {  // fallback: cooperative vector copy (16KB)
    const uint4* src = (const uint4*)(featb + (size_t)b * 8192);
    uint4* dst = (uint4*)sV;
    for (int t = threadIdx.x; t < 1024; t += 256) dst[t] = src[t];
  }
#endif
  __syncthreads();

  const __bf16* wmS = wmT + s * 128;
  __builtin_prefetch(wmS, 0, 3);
  const float* abdB = ABD + (size_t)b * 64 * 384;
  float Cterm = Cb[b * 128 + s];
  float Eterm = Eb[b * 128 + s];

  float total = 0.0f;
#pragma unroll 1
  for (int it = 0; it < 4; ++it) {
    float Av[8], Dv[8];
#pragma unroll
    for (int v = 0; v < 8; ++v) {
      int i = it * 16 + v + 8 * g;
      Av[v] = abdB[i * 384 + s];
      Dv[v] = abdB[i * 384 + 256 + s];
    }
    // A fragments for this row tile: A[i,d] = V[i,d] * wm[s,d]  (bf16)
    bf16x16 afr[4];
#pragma unroll
    for (int kc = 0; kc < 4; ++kc) {
      int d0 = kc * 32 + g * 8;
      int i = it * 16 + mr;
      bf16x8 f0 = *(const bf16x8*)&sV[i * 128 + d0];
      bf16x8 f1 = *(const bf16x8*)&sV[i * 128 + d0 + 16];
      bf16x8 w0 = *(const bf16x8*)(wmS + d0);
      bf16x8 w1 = *(const bf16x8*)(wmS + d0 + 16);
      bf16x16 a;
#pragma unroll
      for (int e = 0; e < 8; ++e) {
        a[e]     = (__bf16)((float)f0[e] * (float)w0[e]);
        a[e + 8] = (__bf16)((float)f1[e] * (float)w1[e]);
      }
      afr[kc] = a;
    }
#pragma unroll 1
    for (int jt = 0; jt < 4; ++jt) {
      f32x8 acc = {};
#pragma unroll
      for (int kc = 0; kc < 4; ++kc) {
        int d0 = kc * 32 + g * 8;
        int j = jt * 16 + mr;
        bf16x8 b0 = *(const bf16x8*)&sV[j * 128 + d0];
        bf16x8 b1 = *(const bf16x8*)&sV[j * 128 + d0 + 16];
        bf16x16 bf;
#pragma unroll
        for (int e = 0; e < 8; ++e) { bf[e] = b0[e]; bf[e + 8] = b1[e]; }
        acc = __builtin_amdgcn_wmma_f32_16x16x32_bf16(false, afr[kc], false, bf,
                                                      (short)0, acc, false, false);
      }
      int j = jt * 16 + mr;
      float Bj = abdB[j * 384 + 128 + s];
#pragma unroll
      for (int v = 0; v < 8; ++v) {
        int i = it * 16 + v + 8 * g;
        float val = acc[v] + Av[v] + Bj + Cterm;
        if (i == j) val += Dv[v] + Eterm;
        total += fmaxf(val, 0.0f);
      }
    }
  }
  // wave32 reduction of per-lane partial sums
#pragma unroll
  for (int off = 16; off > 0; off >>= 1) total += __shfl_xor(total, off, 32);
  if (lane == 0) meanb[b * 128 + s] = total * (1.0f / 4096.0f);
}

// ---------------------------------------------------------------------------
// Final head: out[b] = sum_s relu(mean[b,s]) * out_w[s] + out_b
// ---------------------------------------------------------------------------
__global__ void final_kernel(const float* __restrict__ meanb, const float* __restrict__ outw,
                             const float* __restrict__ outb, float* __restrict__ out) {
  __shared__ float red[128];
  int b = blockIdx.x, s = threadIdx.x;
  red[s] = fmaxf(meanb[b * 128 + s], 0.0f) * outw[s];
  __syncthreads();
  for (int off = 64; off > 0; off >>= 1) {
    if (s < off) red[s] += red[s + off];
    __syncthreads();
  }
  if (s == 0) out[b] = red[0] + outb[0];
}

// ---------------------------------------------------------------------------
extern "C" void kernel_launch(void* const* d_in, const int* in_sizes, int n_in,
                              void* d_out, int out_size, void* d_ws, size_t ws_size,
                              hipStream_t stream) {
  const float* x      = (const float*)d_in[0];
  const float* c1w    = (const float*)d_in[1];
  const float* c1b    = (const float*)d_in[2];
  const float* c2w    = (const float*)d_in[3];
  const float* c2b    = (const float*)d_in[4];
  const float* fcw    = (const float*)d_in[5];
  const float* fcb    = (const float*)d_in[6];
  const float* coeffs = (const float*)d_in[7];
  const float* eqb    = (const float*)d_in[8];
  const float* dgb    = (const float*)d_in[9];
  const float* outw   = (const float*)d_in[10];
  const float* outb   = (const float*)d_in[11];

  char* p = (char*)d_ws;
  auto take = [&](size_t bytes) -> char* {
    char* r = p;
    p += (bytes + 255) & ~(size_t)255;
    return r;
  };
  float*  y1    = (float*)take(1024ull * 32 * 144 * 4);   // conv1 out
  __bf16* act2  = (__bf16*)take(1024ull * 1024 * 2);      // conv2 out (flattened, bf16)
  __bf16* fcwb  = (__bf16*)take(128ull * 1024 * 2);       // fc_w bf16
  float*  featf = (float*)take(1024ull * 128 * 4);        // feat f32
  __bf16* featb = (__bf16*)take(1024ull * 128 * 2);       // feat bf16
  float*  Sbuf  = (float*)take(2048 * 4);
  float*  Tbuf  = (float*)take(2048 * 4);
  __bf16* Xb    = (__bf16*)take(1024ull * 256 * 2);       // [S*v | v^2]
  __bf16* WcT   = (__bf16*)take(384ull * 256 * 2);        // fused A/B/D weights (transposed)
  __bf16* wmT   = (__bf16*)take(128ull * 128 * 2);        // c10+c11 (s-major)
  float*  wC1   = (float*)take(16384 * 4);
  float*  wC2   = (float*)take(16384 * 4);
  float*  wE1   = (float*)take(16384 * 4);
  float*  wE2   = (float*)take(16384 * 4);
  float*  ABD   = (float*)take(1024ull * 384 * 4);        // [A | B | D] per (b,i)
  float*  Cbuf  = (float*)take(2048 * 4);
  float*  Ebuf  = (float*)take(2048 * 4);
  float*  meanb = (float*)take(2048 * 4);

  conv1_kernel<<<18432, 256, 0, stream>>>(x, c1w, c1b, y1);
  conv2_kernel<<<4096, 256, 0, stream>>>(y1, c2w, c2b, act2);
  cvt_bf16_kernel<<<512, 256, 0, stream>>>(fcw, fcwb, 131072);
  prep_coeff_kernel<<<64, 256, 0, stream>>>(coeffs, wmT, WcT, wC1, wC2, wE1, wE2);
  // feat = act2 @ fc_w^T + fc_b   (1024 x 128, K=1024) via WMMA
  gemm_bf16_wmma<<<dim3(64, 8), 32, 0, stream>>>(act2, 1024, fcwb, 1024, fcb,
                                                 featf, 128, featb, 1024);
  stats_kernel<<<8, 256, 0, stream>>>(featf, Sbuf, Tbuf);
  buildX_kernel<<<512, 256, 0, stream>>>(featf, Sbuf, Xb);
  ce_kernel<<<8, 256, 0, stream>>>(Sbuf, Tbuf, wC1, wC2, wE1, wE2, eqb, dgb, Cbuf, Ebuf);
  // [A|B|D] = X @ WcT^T   (1024 x 384, K=256) via WMMA
  gemm_bf16_wmma<<<dim3(64, 24), 32, 0, stream>>>(Xb, 256, WcT, 256, nullptr,
                                                  ABD, 384, nullptr, 256);
  eq_kernel<<<dim3(16, 16), 256, 0, stream>>>(featb, wmT, ABD, Cbuf, Ebuf, meanb);
  final_kernel<<<16, 128, 0, stream>>>(meanb, outw, outb, (float*)d_out);
}
